// cs_net_70574902607892
// MI455X (gfx1250) — matrix operand order repaired and checked by reading, not compiled, not run on hardware
//
#include <hip/hip_runtime.h>
#include <hip/hip_bf16.h>

// ---------------------------------------------------------------------------
// CDNA5 (gfx1250, wave32) implementation.
//  Kernel 1: batched 2-layer LSTM via V_WMMA_F32_16X16X32_F16 (16 samples/wave)
//  Kernel 2: conv + layernorm + MLP head, wave-per-sample, shfl_xor reductions
// ---------------------------------------------------------------------------

typedef __attribute__((ext_vector_type(16))) _Float16 v16h;
typedef __attribute__((ext_vector_type(8)))  _Float16 v8h;
typedef __attribute__((ext_vector_type(8)))  float    v8f;

#define BATCH 65536
#define XSTRIDE 348   // 1*3*116 floats per sample

__device__ __forceinline__ float sigf(float x) {
  return __fdividef(1.0f, 1.0f + __expf(-x));
}
__device__ __forceinline__ float tanhfast(float x) {
  float e = __expf(-2.0f * x);
  return __fdividef(1.0f - e, 1.0f + e);
}

__device__ __forceinline__ float wred(float v) {
#pragma unroll
  for (int m = 16; m > 0; m >>= 1) v += __shfl_xor(v, m, 32);
  return v;
}

// A fragment: 16x32 f16 tile (row-major, stride 32) -> per ISA 16-bit A layout:
// lane L: row = L&15; elems 0..7 = K[base..base+7], elems 8..15 = K[16+base..16+base+7],
// base = (L>=16) ? 8 : 0.  Two aligned 16-byte LDS loads per lane.
__device__ __forceinline__ v16h load_frag_A(const _Float16* __restrict__ tile, int lane) {
  const int row  = lane & 15;
  const int base = (lane >> 4) << 3;
  const _Float16* p = tile + row * 32 + base;
  v8h lo = *(const v8h*)(p);
  v8h hi = *(const v8h*)(p + 16);
  return __builtin_shufflevector(lo, hi, 0, 1, 2, 3, 4, 5, 6, 7,
                                 8, 9, 10, 11, 12, 13, 14, 15);
}

// B fragment: weights W[128][32] (gate-major). Chunk g covers output cols g*16..g*16+15.
// Dense 32x16 16-bit B: lane L elems 0..7 = K[base..base+7], elems 8..15 = K[16+base..16+base+7],
// base = (L>=16) ? 8 : 0.  Two aligned 16-byte LDS loads per lane.
__device__ __forceinline__ v16h load_frag_B(const _Float16* __restrict__ W, int g, int lane) {
  const int col   = lane & 15;
  const int kbase = (lane >> 4) << 3;
  const _Float16* p = W + (g * 16 + col) * 32 + kbase;
  v8h lo = *(const v8h*)(p);
  v8h hi = *(const v8h*)(p + 16);
  return __builtin_shufflevector(lo, hi, 0, 1, 2, 3, 4, 5, 6, 7,
                                 8, 9, 10, 11, 12, 13, 14, 15);
}

// ---------------------------------------------------------------------------
// Kernel 1: LSTM.  block = 256 threads = 8 waves; wave handles 16 samples.
// grid = 65536/128 = 512.  >=4 waves/EU keeps VGPRs <= 256 (no MSB swaps);
// a per-iteration memory clobber stops LICM from hoisting the 24 B-fragments
// into registers (which previously forced scratch spills).
// ---------------------------------------------------------------------------
__global__ __launch_bounds__(256, 4)
void lstm_wmma_kernel(const float* __restrict__ x,
                      const float* __restrict__ Wih0, const float* __restrict__ Whh0,
                      const float* __restrict__ bih0, const float* __restrict__ bhh0,
                      const float* __restrict__ Wih1, const float* __restrict__ Whh1,
                      const float* __restrict__ bih1, const float* __restrict__ bhh1,
                      const float* __restrict__ fcW,  const float* __restrict__ fcb,
                      float* __restrict__ ws6) {
  __shared__ _Float16 sWhh0[128 * 32];
  __shared__ _Float16 sWih1[128 * 32];
  __shared__ _Float16 sWhh1[128 * 32];
  __shared__ float    sWih0[128 * 3];
  __shared__ float    sB0[128], sB1[128];
  __shared__ float    sFcW[96], sFcb[3];
  __shared__ _Float16 sH0[8 * 512];   // per-wave 16x32 h tiles (layer 0)
  __shared__ _Float16 sH1[8 * 512];   // per-wave 16x32 h tiles (layer 1)
  __shared__ float    sPos[8 * 192];  // per-wave 16 samples x 12 pos values

  const int tid  = threadIdx.x;
  const int wv   = tid >> 5;
  const int lane = tid & 31;

  for (int i = tid; i < 128 * 32; i += 256) {
    sWhh0[i] = (_Float16)Whh0[i];
    sWih1[i] = (_Float16)Wih1[i];
    sWhh1[i] = (_Float16)Whh1[i];
  }
  for (int i = tid; i < 128 * 3; i += 256) sWih0[i] = Wih0[i];
  for (int i = tid; i < 128; i += 256) { sB0[i] = bih0[i] + bhh0[i]; sB1[i] = bih1[i] + bhh1[i]; }
  if (tid < 96) sFcW[tid] = fcW[tid];
  if (tid < 3)  sFcb[tid] = fcb[tid];
  __syncthreads();

  _Float16* h0t  = sH0 + wv * 512;
  _Float16* h1t  = sH1 + wv * 512;
  float*    post = sPos + wv * 192;
  const int sbase = blockIdx.x * 128 + wv * 16;

  for (int i = lane; i < 512; i += 32) { h0t[i] = (_Float16)0.0f; h1t[i] = (_Float16)0.0f; }
  // pos = x[:,0,1,104:116]  ->  offset b*348 + 116 + 104 + k
  for (int i = lane; i < 192; i += 32) {
    int s = i / 12, k = i % 12;
    post[i] = x[(size_t)(sbase + s) * XSTRIDE + 220 + k];
  }

  const int col   = lane & 15;
  const int half8 = (lane >> 4) << 3;

  v8f c0[2], c1[2];
#pragma unroll
  for (int j = 0; j < 2; ++j)
#pragma unroll
    for (int r = 0; r < 8; ++r) { c0[j][r] = 0.0f; c1[j][r] = 0.0f; }

#pragma clang loop unroll(disable)
  for (int t = 0; t < 4; ++t) {
    // Force per-iteration reload of LDS-resident weight fragments (prevents
    // LICM from keeping 24 x 8 VGPRs of B-fragments live across the loop).
    asm volatile("" ::: "memory");

    // ------------------ layer 0: gates = bias + xt@Wih0^T (VALU) + h@Whh0^T (WMMA)
    v16h aH0 = load_frag_A(h0t, lane);
#pragma unroll
    for (int j = 0; j < 2; ++j) {      // hidden half j: chunks {j, 2+j, 4+j, 6+j}
      v8f g[4];
#pragma unroll
      for (int q = 0; q < 4; ++q) {
        const int gch  = q * 2 + j;    // q=0->i, 1->f, 2->g, 3->o
        const int gate = gch * 16 + col;
        const float bb = sB0[gate];
        const float w0 = sWih0[gate * 3 + 0];
        const float w1 = sWih0[gate * 3 + 1];
        const float w2 = sWih0[gate * 3 + 2];
        v8f acc;
#pragma unroll
        for (int r = 0; r < 8; ++r) {
          const float* xp = post + (r + half8) * 12 + t * 3;
          acc[r] = bb + w0 * xp[0] + w1 * xp[1] + w2 * xp[2];
        }
        v16h bW = load_frag_B(sWhh0, gch, lane);
        g[q] = __builtin_amdgcn_wmma_f32_16x16x32_f16(false, aH0, false, bW,
                                                      (short)0, acc, false, false);
      }
#pragma unroll
      for (int r = 0; r < 8; ++r) {
        float iv = sigf(g[0][r]), fv = sigf(g[1][r]);
        float gv = tanhfast(g[2][r]), ov = sigf(g[3][r]);
        float cv = fv * c0[j][r] + iv * gv;
        c0[j][r] = cv;
        h0t[(r + half8) * 32 + j * 16 + col] = (_Float16)(ov * tanhfast(cv));
      }
    }

    // ------------------ layer 1: gates = bias + h0_t@Wih1^T + h1@Whh1^T
    v16h aX  = load_frag_A(h0t, lane);
    v16h aH1 = load_frag_A(h1t, lane);
#pragma unroll
    for (int j = 0; j < 2; ++j) {
      v8f g[4];
#pragma unroll
      for (int q = 0; q < 4; ++q) {
        const int gch  = q * 2 + j;
        const int gate = gch * 16 + col;
        v8f acc;
#pragma unroll
        for (int r = 0; r < 8; ++r) acc[r] = sB1[gate];
        v16h bI = load_frag_B(sWih1, gch, lane);
        acc = __builtin_amdgcn_wmma_f32_16x16x32_f16(false, aX, false, bI,
                                                     (short)0, acc, false, false);
        v16h bH = load_frag_B(sWhh1, gch, lane);
        g[q] = __builtin_amdgcn_wmma_f32_16x16x32_f16(false, aH1, false, bH,
                                                      (short)0, acc, false, false);
      }
#pragma unroll
      for (int r = 0; r < 8; ++r) {
        float iv = sigf(g[0][r]), fv = sigf(g[1][r]);
        float gv = tanhfast(g[2][r]), ov = sigf(g[3][r]);
        float cv = fv * c1[j][r] + iv * gv;
        c1[j][r] = cv;
        h1t[(r + half8) * 32 + j * 16 + col] = (_Float16)(ov * tanhfast(cv));
      }
    }
  }

  // epilogue: pos_p = h1_final @ fcW^T + fcb ; vel = (pos_p - pos_r)*10
  if (lane < 16) {
    const size_t s = (size_t)sbase + lane;
#pragma unroll
    for (int j = 0; j < 3; ++j) {
      float p = sFcb[j];
      for (int k = 0; k < 32; ++k) p += sFcW[j * 32 + k] * (float)h1t[lane * 32 + k];
      ws6[s * 6 + j]     = p;
      ws6[s * 6 + 3 + j] = (p - post[lane * 12 + 9 + j]) * 10.0f;
    }
  }
}

// ---------------------------------------------------------------------------
// Kernel 2: conv + LN + MLP head.  block = 128 threads = 4 waves; wave = 1 sample.
// grid = 65536/4 = 16384.
// ---------------------------------------------------------------------------
__global__ __launch_bounds__(128, 4)
void head_kernel(const float* __restrict__ x,
                 const float* __restrict__ cv2W, const float* __restrict__ cv2b,
                 const float* __restrict__ cv2g, const float* __restrict__ cv2be,
                 const float* __restrict__ cv3W, const float* __restrict__ cv3b,
                 const float* __restrict__ cv3g, const float* __restrict__ cv3be,
                 const float* __restrict__ f1W,  const float* __restrict__ f1b,
                 const float* __restrict__ f1g,  const float* __restrict__ f1be,
                 const float* __restrict__ f2W,  const float* __restrict__ f2b,
                 const float* __restrict__ f2g,  const float* __restrict__ f2be,
                 const float* __restrict__ f3W,  const float* __restrict__ f3b,
                 const float* __restrict__ f3g,  const float* __restrict__ f3be,
                 const float* __restrict__ ws6,  float* __restrict__ out) {
  __shared__ float sC2W[60], sC2b[4], sC2g[408], sC2be[408];
  __shared__ float sC3W[4], sC3g[102], sC3be[102];
  __shared__ float sC3b;
  __shared__ float sF1W[3456], sF1b[32], sF1g[32], sF1be[32];
  __shared__ float sF2W[1024], sF2b[32], sF2g[32], sF2be[32];
  __shared__ float sF3W[3264], sF3b[102], sF3g[102], sF3be[102];
  __shared__ float sCsi[4][306];
  __shared__ float sOut2[4][408];
  __shared__ float sOut4[4][108];
  __shared__ float sVa[4][32], sVb[4][32];

  const int tid  = threadIdx.x;
  const int wv   = tid >> 5;
  const int lane = tid & 31;

  for (int i = tid; i < 60;   i += 128) sC2W[i] = cv2W[i];
  for (int i = tid; i < 4;    i += 128) { sC2b[i] = cv2b[i]; sC3W[i] = cv3W[i]; }
  for (int i = tid; i < 408;  i += 128) { sC2g[i] = cv2g[i]; sC2be[i] = cv2be[i]; }
  for (int i = tid; i < 102;  i += 128) {
    sC3g[i] = cv3g[i]; sC3be[i] = cv3be[i];
    sF3b[i] = f3b[i];  sF3g[i]  = f3g[i];  sF3be[i] = f3be[i];
  }
  for (int i = tid; i < 3456; i += 128) sF1W[i] = f1W[i];
  for (int i = tid; i < 1024; i += 128) sF2W[i] = f2W[i];
  for (int i = tid; i < 3264; i += 128) sF3W[i] = f3W[i];
  for (int i = tid; i < 32;   i += 128) {
    sF1b[i] = f1b[i]; sF1g[i] = f1g[i]; sF1be[i] = f1be[i];
    sF2b[i] = f2b[i]; sF2g[i] = f2g[i]; sF2be[i] = f2be[i];
  }
  if (tid == 0) sC3b = cv3b[0];
  __syncthreads();

  const size_t b = (size_t)blockIdx.x * 4 + wv;
  float* csi = sCsi[wv];
  float* o2  = sOut2[wv];
  float* o4  = sOut4[wv];

  // csi = x[:, :, :, 1:103]
  const float* xb = x + b * XSTRIDE;
  for (int i = lane; i < 306; i += 32) {
    int r = i / 102, c = i % 102;
    csi[i] = xb[r * 116 + 1 + c];
  }

  // conv2 (3x5 kernel, pad_w=2) -> 4x102 outputs, raw (LN applied lazily)
  float s2 = 0.0f, ss2 = 0.0f;
  for (int o = lane; o < 408; o += 32) {
    int ch = o / 102, w = o % 102;
    float acc = sC2b[ch];
#pragma unroll
    for (int r = 0; r < 3; ++r)
#pragma unroll
      for (int dw = 0; dw < 5; ++dw) {
        int cc = w + dw - 2;
        if (cc >= 0 && cc < 102) acc += sC2W[ch * 15 + r * 5 + dw] * csi[r * 102 + cc];
      }
    o2[o] = acc; s2 += acc; ss2 += acc * acc;
  }
  s2 = wred(s2); ss2 = wred(ss2);
  float m2 = s2 / 408.0f;
  float r2 = rsqrtf(ss2 / 408.0f - m2 * m2 + 1e-5f);

  // conv3 (1x1 over 4 ch) on relu(LN(out2)), then LN(102) + relu -> out4[0:102]
  float o3v[4]; float s3 = 0.0f, ss3 = 0.0f;
#pragma unroll
  for (int q = 0; q < 4; ++q) {
    int w = lane + 32 * q;
    float acc = 0.0f;
    if (w < 102) {
      acc = sC3b;
#pragma unroll
      for (int ch = 0; ch < 4; ++ch) {
        float v = (o2[ch * 102 + w] - m2) * r2 * sC2g[ch * 102 + w] + sC2be[ch * 102 + w];
        acc += sC3W[ch] * fmaxf(v, 0.0f);
      }
      s3 += acc; ss3 += acc * acc;
    }
    o3v[q] = acc;
  }
  s3 = wred(s3); ss3 = wred(ss3);
  float m3 = s3 / 102.0f;
  float r3 = rsqrtf(ss3 / 102.0f - m3 * m3 + 1e-5f);
#pragma unroll
  for (int q = 0; q < 4; ++q) {
    int w = lane + 32 * q;
    if (w < 102) o4[w] = fmaxf((o3v[q] - m3) * r3 * sC3g[w] + sC3be[w], 0.0f);
  }
  if (lane < 6) o4[102 + lane] = ws6[b * 6 + lane];  // pos_p(3), vel(3)

  // f1: 108 -> 32, LN(32), relu    (lane == output neuron)
  float h1 = sF1b[lane];
  for (int k = 0; k < 108; ++k) h1 += sF1W[lane * 108 + k] * o4[k];
  float sm = wred(h1) / 32.0f;
  float sv = wred(h1 * h1) / 32.0f - sm * sm;
  h1 = fmaxf((h1 - sm) * rsqrtf(sv + 1e-5f) * sF1g[lane] + sF1be[lane], 0.0f);
  sVa[wv][lane] = h1;

  // f2: 32 -> 32, LN(32), relu
  float h2 = sF2b[lane];
  for (int k = 0; k < 32; ++k) h2 += sF2W[lane * 32 + k] * sVa[wv][k];
  sm = wred(h2) / 32.0f;
  sv = wred(h2 * h2) / 32.0f - sm * sm;
  h2 = fmaxf((h2 - sm) * rsqrtf(sv + 1e-5f) * sF2g[lane] + sF2be[lane], 0.0f);
  sVb[wv][lane] = h2;

  // f3: 32 -> 102, LN(102), sigmoid
  float ov[4]; float sf = 0.0f, sfs = 0.0f;
#pragma unroll
  for (int q = 0; q < 4; ++q) {
    int w = lane + 32 * q;
    float acc = 0.0f;
    if (w < 102) {
      acc = sF3b[w];
      for (int k = 0; k < 32; ++k) acc += sF3W[w * 32 + k] * sVb[wv][k];
      sf += acc; sfs += acc * acc;
    }
    ov[q] = acc;
  }
  sf = wred(sf); sfs = wred(sfs);
  float mf = sf / 102.0f;
  float rf = rsqrtf(sfs / 102.0f - mf * mf + 1e-5f);
#pragma unroll
  for (int q = 0; q < 4; ++q) {
    int w = lane + 32 * q;
    if (w < 102) out[b * 102 + w] = sigf((ov[q] - mf) * rf * sF3g[w] + sF3be[w]);
  }
}

// ---------------------------------------------------------------------------
extern "C" void kernel_launch(void* const* d_in, const int* in_sizes, int n_in,
                              void* d_out, int out_size, void* d_ws, size_t ws_size,
                              hipStream_t stream) {
  const float* x    = (const float*)d_in[0];
  const float* Wih0 = (const float*)d_in[1];
  const float* Whh0 = (const float*)d_in[2];
  const float* bih0 = (const float*)d_in[3];
  const float* bhh0 = (const float*)d_in[4];
  const float* Wih1 = (const float*)d_in[5];
  const float* Whh1 = (const float*)d_in[6];
  const float* bih1 = (const float*)d_in[7];
  const float* bhh1 = (const float*)d_in[8];
  const float* fcW  = (const float*)d_in[9];
  const float* fcb  = (const float*)d_in[10];
  const float* cv2W = (const float*)d_in[11];
  const float* cv2b = (const float*)d_in[12];
  const float* cv2g = (const float*)d_in[13];
  const float* cv2be= (const float*)d_in[14];
  const float* cv3W = (const float*)d_in[15];
  const float* cv3b = (const float*)d_in[16];
  const float* cv3g = (const float*)d_in[17];
  const float* cv3be= (const float*)d_in[18];
  const float* f1W  = (const float*)d_in[19];
  const float* f1b  = (const float*)d_in[20];
  const float* f1g  = (const float*)d_in[21];
  const float* f1be = (const float*)d_in[22];
  const float* f2W  = (const float*)d_in[23];
  const float* f2b  = (const float*)d_in[24];
  const float* f2g  = (const float*)d_in[25];
  const float* f2be = (const float*)d_in[26];
  const float* f3W  = (const float*)d_in[27];
  const float* f3b  = (const float*)d_in[28];
  const float* f3g  = (const float*)d_in[29];
  const float* f3be = (const float*)d_in[30];

  float* ws6 = (float*)d_ws;  // 65536 * 6 floats: pos_p(3) + vel(3)

  lstm_wmma_kernel<<<BATCH / 128, 256, 0, stream>>>(
      x, Wih0, Whh0, bih0, bhh0, Wih1, Whh1, bih1, bhh1, fcW, fcb, ws6);

  head_kernel<<<BATCH / 4, 128, 0, stream>>>(
      x, cv2W, cv2b, cv2g, cv2be, cv3W, cv3b, cv3g, cv3be,
      f1W, f1b, f1g, f1be, f2W, f2b, f2g, f2be, f3W, f3b, f3g, f3be,
      ws6, (float*)d_out);
}
